// GraphConvolution_48292612276717
// MI455X (gfx1250) — matrix operand order repaired and checked by reading, not compiled, not run on hardware
//
#include <hip/hip_runtime.h>

// GraphConvolution, MI455X / gfx1250, wave32 + WMMA bf16 (f32 accumulate)
// + double-buffered GLOBAL_LOAD_ASYNC_TO_LDS pipeline.
//
// ALPHA==1.0  =>  m_norm == adj_norm exactly; the 2048^3 adj@adj matmul is dead.
// Pipeline:
//   1) r_inv[m] = rsqrt(colsum(adj))                                  [16 MB read]
//   2) adj_sw   = bf16(adj[n][m] * r_inv[n]) pre-swizzled into WMMA
//      A-fragment order, 8KB chunks per (32k x 128m) tile             [one-time]
//   3) X_comb   = bf16(x @ W) written in WMMA B-fragment order,
//      8KB chunks per (32k x 128col) tile, col = g*128 + d            [1.1 GFLOP]
//   4) out = relu(r_inv[m] * adj_sw^T-GEMM + bias): M=2048, N=2048
//      (16 batches x 128), K=2048, 17.2 GFLOP — hot loop is pure
//      async-copy -> ds_load_b128 -> v_wmma, zero conversion VALU.

typedef __attribute__((ext_vector_type(16))) __bf16 v16bf;
typedef __attribute__((ext_vector_type(8)))  float  v8f;
typedef int v4i_ __attribute__((vector_size(16)));

#define N_NODES 2048
#define DIN     128
#define DOUT    128

// f32 -> bf16 round-to-nearest-even (finite inputs only)
__device__ __forceinline__ unsigned short f2bf(float f) {
  unsigned u = __float_as_uint(f);
  return (unsigned short)((u + 0x7FFFu + ((u >> 16) & 1u)) >> 16);
}

// ---- CDNA5 async global->LDS copy (ASYNCcnt-tracked), 16B per lane ----------
typedef __attribute__((address_space(1))) v4i_ gv4i;
typedef __attribute__((address_space(3))) v4i_ lv4i;
typedef __attribute__((address_space(1))) void gvoid;
typedef __attribute__((address_space(3))) void lvoid;

__device__ __forceinline__ void async_b128(const unsigned short* g,
                                           unsigned short* l) {
#if __has_builtin(__builtin_amdgcn_global_load_async_to_lds_b128)
  __builtin_amdgcn_global_load_async_to_lds_b128((gv4i*)g, (lv4i*)l, 0, 0);
#else
  asm volatile("global_load_async_to_lds_b128 %0, %1, off"
               :
               : "v"((lvoid*)l), "v"((gvoid*)g)
               : "memory");
#endif
}

#if __has_builtin(__builtin_amdgcn_s_wait_asynccnt)
#define WAIT_ASYNC(n) __builtin_amdgcn_s_wait_asynccnt(n)
#else
#define WAIT_ASYNC(n) asm volatile("s_wait_asynccnt %0" ::"i"(n) : "memory")
#endif

// ---------------------------------------------------------------------------
// Kernel 1: r_inv[m] = rsqrt(sum_n adj[n][m])
// ---------------------------------------------------------------------------
__global__ __launch_bounds__(256) void gcn_colsum_rsqrt(
    const float* __restrict__ adj, float* __restrict__ r_inv) {
  const int m = blockIdx.x * 256 + threadIdx.x;
  float s = 0.f;
  for (int n = 0; n < N_NODES; ++n) s += adj[(size_t)n * N_NODES + m];
  r_inv[m] = rsqrtf(s);
}

// ---------------------------------------------------------------------------
// Fragment layouts (CDNA5 ISA 7.12.2, 16-bit, wave32):
//   A 16x32: lane = (m&15) + 16*((k&15)>=8), elem i = (k&7) + 8*(k>>4)
//   B 32x16: lane = (n&15) + 16*((k&15)>=8), elem i = (k&7) + 8*(k>>4)
// Chunk = 32(K) x 128(M or N) = 4096 bf16 = 8KB, internally [subtile][lane][i].
// ---------------------------------------------------------------------------

// Kernel 2: adj_sw[(kblk*16+mblk)*4096 + frag] = bf16(adj[n][m] * r_inv[n])
__global__ __launch_bounds__(256) void gcn_adj_swizzle(
    const float* __restrict__ adj, const float* __restrict__ r_inv,
    unsigned short* __restrict__ adj_sw) {
  __shared__ __align__(32) unsigned short Abuf[4096];
  const int t    = threadIdx.x;
  const int mblk = blockIdx.x;  // 0..15
  const int kblk = blockIdx.y;  // 0..63
  const int kk   = t >> 3;      // 0..31 = n & 31
  const int n    = kblk * 32 + kk;
  const int mb   = mblk * 128 + (t & 7) * 16;
  const float rn = r_inv[n];
  const float4* src = (const float4*)(adj + (size_t)n * N_NODES + mb);
  const int s  = t & 7;
  const int kh = ((kk & 15) >= 8) ? 16 : 0;
  const int i  = (kk & 7) + ((kk >> 4) << 3);
#pragma unroll
  for (int v4 = 0; v4 < 4; ++v4) {
    const float4 f = src[v4];
    Abuf[s * 512 + (v4 * 4 + 0 + kh) * 16 + i] = f2bf(f.x * rn);
    Abuf[s * 512 + (v4 * 4 + 1 + kh) * 16 + i] = f2bf(f.y * rn);
    Abuf[s * 512 + (v4 * 4 + 2 + kh) * 16 + i] = f2bf(f.z * rn);
    Abuf[s * 512 + (v4 * 4 + 3 + kh) * 16 + i] = f2bf(f.w * rn);
  }
  __syncthreads();
  uint4* dst = (uint4*)(adj_sw + ((size_t)(kblk * 16 + mblk)) * 4096);
  const uint4* sb = (const uint4*)Abuf;
  dst[t * 2]     = sb[t * 2];
  dst[t * 2 + 1] = sb[t * 2 + 1];
}

// Kernel 3: projection X_comb = bf16(x @ W), written in B-fragment chunks
__global__ __launch_bounds__(256) void gcn_project(
    const float* __restrict__ x, const float* __restrict__ w,
    unsigned short* __restrict__ xc_sw) {
  __shared__ __align__(32) unsigned short Asub[8][32][16];
  __shared__ __align__(32) unsigned short Bsub[8][32][16];
  const int t    = threadIdx.x;
  const int lane = t & 31;
  const int wid  = t >> 5;
  const int wm   = wid & 3;
  const int wn   = wid >> 2;
  const int row0 = blockIdx.x * 128;

  v8f acc[2][4] = {};
  for (int k0 = 0; k0 < DIN; k0 += 32) {
    __syncthreads();
    {  // stage A: 128 rows x 32 k of x, fragment order
      const int m  = t >> 1;
      const int kb = (t & 1) * 16;
      const float* src = x + (size_t)(row0 + m) * DIN + k0 + kb;
      const int s = m >> 4, mr = m & 15;
#pragma unroll
      for (int ii = 0; ii < 16; ++ii) {
        const int k  = kb + ii;
        const int kh = ((k & 15) >= 8) ? 1 : 0;
        const int i  = (k & 7) + ((k >> 4) << 3);
        Asub[s][mr + 16 * kh][i] = f2bf(src[ii]);
      }
    }
    {  // stage B: 32 k x 128 d of W, fragment order
      const int k  = t >> 3;
      const int nb = (t & 7) * 16;
      const float* src = w + (size_t)(k0 + k) * DOUT + nb;
      const int j = t & 7, i = (k & 7) + ((k >> 4) << 3), lb = ((k & 15) >= 8) * 16;
#pragma unroll
      for (int ii = 0; ii < 16; ++ii) Bsub[j][lb + ii][i] = f2bf(src[ii]);
    }
    __syncthreads();
    v16bf a[2], b[4];
#pragma unroll
    for (int p = 0; p < 2; ++p) a[p] = *(const v16bf*)&Asub[wm * 2 + p][lane][0];
#pragma unroll
    for (int q = 0; q < 4; ++q) b[q] = *(const v16bf*)&Bsub[wn * 4 + q][lane][0];
#pragma unroll
    for (int p = 0; p < 2; ++p)
#pragma unroll
      for (int q = 0; q < 4; ++q)
        acc[p][q] = __builtin_amdgcn_wmma_f32_16x16x32_bf16(
            false, a[p], false, b[q], (short)0, acc[p][q], false, false);
  }
  // epilogue: write bf16 into B-fragment-swizzled chunks:
  // chunk = (n>>5)*16 + g ; off = (d>>4)*512 + (((kk&15)>=8)*16 + (d&15))*16 + (kk&7) + 8*(kk>>4)
  const int Ml = (lane >> 4) * 8, Nl = lane & 15;
#pragma unroll
  for (int p = 0; p < 2; ++p)
#pragma unroll
    for (int q = 0; q < 4; ++q)
#pragma unroll
      for (int r = 0; r < 8; ++r) {
        const int row = row0 + (wm * 2 + p) * 16 + Ml + r;  // = g*2048 + n
        const int d   = (wn * 4 + q) * 16 + Nl;
        const int g   = row >> 11;
        const int n   = row & (N_NODES - 1);
        const int kk  = n & 31;
        const size_t chunk = (size_t)((n >> 5) * 16 + g) * 4096;
        const int off = (d >> 4) * 512 + ((((kk & 15) >= 8) * 16) + (d & 15)) * 16 + (kk & 7) + ((kk >> 4) << 3);
        xc_sw[chunk + off] = f2bf(acc[p][q][r]);
      }
}

// Kernel 4: aggregation GEMM, double-buffered async-to-LDS pipeline
__global__ __launch_bounds__(256) void gcn_aggregate(
    const unsigned short* __restrict__ adj_sw,
    const unsigned short* __restrict__ xc_sw,
    const float* __restrict__ r_inv, const float* __restrict__ bias,
    float* __restrict__ out) {
  __shared__ __align__(32) unsigned short Abuf[2][4096];
  __shared__ __align__(32) unsigned short Bbuf[2][4096];
  const int t    = threadIdx.x;
  const int lane = t & 31;
  const int wid  = t >> 5;
  const int wm   = wid & 3;
  const int wn   = wid >> 2;
  const int mblk = blockIdx.x;  // 0..15 : m tile
  const int cblk = blockIdx.y;  // 0..15 : col tile (== batch g)

  // each thread async-copies 32B of the A chunk and 32B of the B chunk
  const unsigned short* gA = adj_sw + (size_t)mblk * 4096 + t * 16;
  const unsigned short* gB = xc_sw + (size_t)cblk * 4096 + t * 16;
  auto issue = [&](int kb, int buf) {
    const unsigned short* ga = gA + (size_t)kb * (16 * 4096);
    const unsigned short* gb = gB + (size_t)kb * (16 * 4096);
    unsigned short* la = &Abuf[buf][t * 16];
    unsigned short* lb = &Bbuf[buf][t * 16];
    async_b128(ga, la);
    async_b128(ga + 8, la + 8);
    async_b128(gb, lb);
    async_b128(gb + 8, lb + 8);
  };

  issue(0, 0);
  v8f acc[2][4] = {};
  for (int kb = 0; kb < 64; ++kb) {
    const int cur = kb & 1;
    if (kb + 1 < 64) {
      issue(kb + 1, cur ^ 1);
      WAIT_ASYNC(4);  // the 4 just-issued remain; current buffer is complete
    } else {
      WAIT_ASYNC(0);
    }
    __syncthreads();
    v16bf a[2], b[4];
#pragma unroll
    for (int p = 0; p < 2; ++p)
      a[p] = *(const v16bf*)&Abuf[cur][(wm * 2 + p) * 512 + lane * 16];
#pragma unroll
    for (int q = 0; q < 4; ++q)
      b[q] = *(const v16bf*)&Bbuf[cur][(wn * 4 + q) * 512 + lane * 16];
#pragma unroll
    for (int p = 0; p < 2; ++p)
#pragma unroll
      for (int q = 0; q < 4; ++q)
        acc[p][q] = __builtin_amdgcn_wmma_f32_16x16x32_bf16(
            false, a[p], false, b[q], (short)0, acc[p][q], false, false);
    __syncthreads();  // protect buffer cur before it is refilled at kb+2
  }

  // epilogue: r_inv[m] * acc + bias, relu, scatter to out[g][m][d]
  const int Ml = (lane >> 4) * 8, Nl = lane & 15;
#pragma unroll
  for (int p = 0; p < 2; ++p)
#pragma unroll
    for (int q = 0; q < 4; ++q) {
      const int col = cblk * 128 + (wn * 4 + q) * 16 + Nl;
      const int g = col >> 7, d = col & 127;
      const float bd = bias[d];
#pragma unroll
      for (int r = 0; r < 8; ++r) {
        const int m = mblk * 128 + (wm * 2 + p) * 16 + Ml + r;
        const float v = fmaxf(r_inv[m] * acc[p][q][r] + bd, 0.f);
        out[((size_t)g * N_NODES + m) * DOUT + d] = v;
      }
    }
}

// ---------------------------------------------------------------------------
extern "C" void kernel_launch(void* const* d_in, const int* in_sizes, int n_in,
                              void* d_out, int out_size, void* d_ws,
                              size_t ws_size, hipStream_t stream) {
  const float* adj  = (const float*)d_in[0];  // [2048,2048]
  const float* x    = (const float*)d_in[1];  // [2,8,2048,128]
  const float* w    = (const float*)d_in[2];  // [128,128]
  const float* bias = (const float*)d_in[3];  // [128]
  float* out = (float*)d_out;                 // [2,8,2048,128]

  float* r_inv = (float*)d_ws;                                        // 8 KB
  unsigned short* adj_sw = (unsigned short*)((char*)d_ws + 8192);     // 8 MB
  unsigned short* xc_sw  = adj_sw + (size_t)N_NODES * N_NODES;        // 8 MB

  gcn_colsum_rsqrt<<<N_NODES / 256, 256, 0, stream>>>(adj, r_inv);
  gcn_adj_swizzle<<<dim3(16, 64), 256, 0, stream>>>(adj, r_inv, adj_sw);
  gcn_project<<<(16 * N_NODES) / 128, 256, 0, stream>>>(x, w, xc_sw);
  gcn_aggregate<<<dim3(16, 16), 256, 0, stream>>>(adj_sw, xc_sw, r_inv, bias,
                                                  out);
}
